// IPANetPredictor_85383949845225
// MI455X (gfx1250) — matrix operand order repaired and checked by reading, not compiled, not run on hardware
//
#include <hip/hip_runtime.h>
#include <math.h>

// ---------------------------------------------------------------------------
// MI455X (gfx1250) implementation of the IPA network forward pass.
// All GEMMs run on v_wmma_f32_16x16x32_bf16 (bf16 A/B, f32 accumulate).
// Tile staging uses CDNA5 async global->LDS copies (ASYNCcnt, via inline asm
// `global_load_async_to_lds_b128` + `s_wait_asynccnt`); falls back to
// VGPR-staged loads off-device. Softmax/layernorm/frames stay in f32.
// ---------------------------------------------------------------------------

typedef __attribute__((ext_vector_type(16))) __bf16 v16bf;
typedef __attribute__((ext_vector_type(8)))  float  v8f;

union ABfrag { uint4 q[2]; v16bf v; };          // 32 bytes
union U32B   { unsigned int u; __bf16 b[2]; };

__device__ __forceinline__ v8f wmma_bf16(v16bf a, v16bf b, v8f c) {
#if defined(__HIP_DEVICE_COMPILE__)
  return __builtin_amdgcn_wmma_f32_16x16x32_bf16(false, a, false, b, (short)0, c,
                                                 false, false);
#else
  return c;
#endif
}

// ---- CDNA5 async global->LDS (tracked by ASYNCcnt), via inline asm ---------
#if defined(__HIP_DEVICE_COMPILE__) && defined(__gfx1250__)
#  define HAS_ASYNC_LDS 1
__device__ __forceinline__ void async_copy16(const void* g, void* l) {
  // per-lane 16B DMA: LDS[l] = GLOBAL[g]; increments ASYNCcnt.
  // Generic LDS address truncates to the LDS byte offset (ISA 10.2 aperture).
  const unsigned int lds_off = (unsigned int)(unsigned long long)l;
  asm volatile("global_load_async_to_lds_b128 %0, %1, off"
               :: "v"(lds_off), "v"(g)
               : "memory");
}
__device__ __forceinline__ void async_wait0() {
  asm volatile("s_wait_asynccnt 0x0" ::: "memory");
}
#else
#  define HAS_ASYNC_LDS 0
#endif

// ---------------------------------------------------------------------------
// Generic batched GEMM: C[m,n] = act(alpha * sum_k A[m,k]*B[k,n] + bias[n])
//   A : bf16, row-major (M x K), row stride lda, batch stride sA
//   B : bf16. BKxN=true  -> stored (K x N), row stride ldb (activations)
//             BKxN=false -> stored (N x K), row stride ldb (A*B^T; weights are
//                           pre-transposed so they take this fast path)
//   C : OutT (float or bf16), row stride ldc, batch stride sC
//   C2: optional bf16 mirror of C (same ldc/sC)
// K must be a multiple of 32 (true for every GEMM in this model).
// Block tile 64x64, 8 waves: wave w -> (wm = w&3) 16-row, (wn = w>>2) 32-col.
// ---------------------------------------------------------------------------
template<bool BKxN, typename OutT, int ACT>
__global__ __launch_bounds__(256) void k_gemm(
    int M, int N, int K,
    const __bf16* __restrict__ A, int lda, long sA,
    const __bf16* __restrict__ B, int ldb, long sB,
    OutT* __restrict__ C, int ldc, long sC,
    const float* __restrict__ bias,
    __bf16* __restrict__ C2,
    float alpha)
{
  __shared__ __bf16 As[2][64 * 40];   // row stride 40 (80B) -> conflict-free
  __shared__ __bf16 Bs[2][64 * 40];   // stored [n][k]

  const int bz = blockIdx.z;
  A += (long)bz * sA;
  B += (long)bz * sB;
  C += (long)bz * sC;
  __bf16* C2l = C2 ? (C2 + (long)bz * sC) : nullptr;

  const int n0 = blockIdx.x * 64;
  const int m0 = blockIdx.y * 64;
  const int t  = threadIdx.x;

  auto loadA = [&](int buf, int k0) {
    const int r  = t >> 2;
    const int c0 = (t & 3) * 8;
    const int gm = m0 + r;
    const __bf16* src = A + (long)gm * lda + k0 + c0;
    __bf16* dstp = &As[buf][r * 40 + c0];
#if HAS_ASYNC_LDS
    if (gm < M) {
      async_copy16(src, dstp);
    } else {
      unsigned int* d32 = (unsigned int*)dstp;
#pragma unroll
      for (int i = 0; i < 4; ++i) d32[i] = 0u;
    }
#else
    unsigned int* d32 = (unsigned int*)dstp;
#pragma unroll
    for (int i = 0; i < 4; ++i)
      d32[i] = (gm < M) ? *(const unsigned int*)(src + 2 * i) : 0u;
#endif
  };

  auto loadB = [&](int buf, int k0) {
    if constexpr (BKxN) {
      // activation B stored (K x N): transpose while staging
      const int kk = t >> 3;            // 0..31
      const int c0 = (t & 7) * 8;       // 0..56
      const __bf16* src = B + (long)(k0 + kk) * ldb + n0 + c0;
#pragma unroll
      for (int i = 0; i < 4; ++i) {
        const int gn = n0 + c0 + 2 * i;          // N is always even
        U32B u; u.u = (gn < N) ? *(const unsigned int*)(src + 2 * i) : 0u;
        Bs[buf][(c0 + 2 * i + 0) * 40 + kk] = u.b[0];
        Bs[buf][(c0 + 2 * i + 1) * 40 + kk] = u.b[1];
      }
    } else {
      const int r  = t >> 2;
      const int c0 = (t & 3) * 8;
      const int gn = n0 + r;
      const __bf16* src = B + (long)gn * ldb + k0 + c0;
      __bf16* dstp = &Bs[buf][r * 40 + c0];
#if HAS_ASYNC_LDS
      if (gn < N) {
        async_copy16(src, dstp);
      } else {
        unsigned int* d32 = (unsigned int*)dstp;
#pragma unroll
        for (int i = 0; i < 4; ++i) d32[i] = 0u;
      }
#else
      unsigned int* d32 = (unsigned int*)dstp;
#pragma unroll
      for (int i = 0; i < 4; ++i)
        d32[i] = (gn < N) ? *(const unsigned int*)(src + 2 * i) : 0u;
#endif
    }
  };

  const int wid   = t >> 5;
  const int lane  = t & 31;
  const int wm    = wid & 3;
  const int wn    = wid >> 2;
  const int l16   = lane & 15;
  const int khalf = lane >> 4;

  v8f acc[2];
  acc[0] = (v8f){0.f,0.f,0.f,0.f,0.f,0.f,0.f,0.f};
  acc[1] = acc[0];

  const int nk = K / 32;
  loadA(0, 0);
  loadB(0, 0);
  for (int kk = 0; kk < nk; ++kk) {
#if HAS_ASYNC_LDS
    async_wait0();                 // my async copies have landed in LDS
#endif
    __syncthreads();               // everyone's copies have landed
    const int cur = kk & 1;
    if (kk + 1 < nk) { loadA(cur ^ 1, (kk + 1) * 32); loadB(cur ^ 1, (kk + 1) * 32); }
    if constexpr (BKxN) {
      if (kk + 2 < nk)   // hint next activation tile into cache
        __builtin_prefetch(B + (long)((kk + 2) * 32 + (t >> 3)) * ldb + n0, 0, 1);
    }
    ABfrag aU, b0U, b1U;
    const __bf16* ar = &As[cur][(wm * 16 + l16) * 40 + khalf * 8];
    aU.q[0] = *(const uint4*)(ar);
    aU.q[1] = *(const uint4*)(ar + 16);
    const __bf16* b0 = &Bs[cur][(wn * 32 + l16) * 40 + khalf * 8];
    b0U.q[0] = *(const uint4*)(b0);
    b0U.q[1] = *(const uint4*)(b0 + 16);
    const __bf16* b1 = &Bs[cur][(wn * 32 + 16 + l16) * 40 + khalf * 8];
    b1U.q[0] = *(const uint4*)(b1);
    b1U.q[1] = *(const uint4*)(b1 + 16);
    acc[0] = wmma_bf16(aU.v, b0U.v, acc[0]);
    acc[1] = wmma_bf16(aU.v, b1U.v, acc[1]);
  }

#pragma unroll
  for (int sub = 0; sub < 2; ++sub) {
    const int gn = n0 + wn * 32 + sub * 16 + l16;
    if (gn >= N) continue;
    const float bb = bias ? bias[gn] : 0.f;
#pragma unroll
    for (int v = 0; v < 8; ++v) {
      const int gm = m0 + wm * 16 + khalf * 8 + v;
      if (gm >= M) continue;
      float val = acc[sub][v] * alpha + bb;
      if (ACT == 1) val = fmaxf(val, 0.f);
      const long o = (long)gm * ldc + gn;
      C[o] = (OutT)val;
      if (C2l) C2l[o] = (__bf16)val;
    }
  }
}

// ---------------------------------------------------------------------------
// Elementwise / small kernels
// ---------------------------------------------------------------------------
__device__ __forceinline__ void norml3(float* v, float eps) {
  float nn = sqrtf(v[0]*v[0] + v[1]*v[1] + v[2]*v[2]) + eps;
  v[0] /= nn; v[1] /= nn; v[2] /= nn;
}
__device__ __forceinline__ void cross3(const float* a, const float* b, float* c) {
  c[0] = a[1]*b[2] - a[2]*b[1];
  c[1] = a[2]*b[0] - a[0]*b[2];
  c[2] = a[0]*b[1] - a[1]*b[0];
}
__device__ __forceinline__ float dot3(const float* a, const float* b) {
  return a[0]*b[0] + a[1]*b[1] + a[2]*b[2];
}

// frames + dihedral feats + posenc + s init. One block per residue (128 thr).
__global__ __launch_bounds__(128) void k_prep(
    const float* __restrict__ xaa,   // (B,L,20)
    const float* __restrict__ xpos,  // (B,L,4,3)
    const int* __restrict__ aamask, const int* __restrict__ seqmask,
    const float* __restrict__ Waa, const float* __restrict__ baa,
    const float* __restrict__ Wd,  const float* __restrict__ bd,
    const float* __restrict__ maskemb, const float* __restrict__ pademb,
    float* __restrict__ s, __bf16* __restrict__ sb,
    float* __restrict__ rot, float* __restrict__ trans, int n)
{
  const int i = blockIdx.x % n;
  const int b = blockIdx.x / n;
  const long ri = (long)b * n + i;
  const int t = threadIdx.x;
  __shared__ float df[6];
  __shared__ float aabuf[20];

  if (t < 20) aabuf[t] = xaa[ri * 20 + t];

  if (t < 3) {                       // dihedral D for this residue (3 values)
    const int k = 3 * i - 1 + t;
    float D = 0.f;
    if (k >= 0 && k <= 3 * n - 4) {
      float X[4][3];
      for (int m = 0; m < 4; ++m)
        for (int d = 0; d < 3; ++d)
          X[m][d] = xpos[((long)b * n + (k + m) / 3) * 12 + ((k + m) % 3) * 3 + d];
      float u2[3], u1[3], u0[3], n2[3], n1[3];
      for (int d = 0; d < 3; ++d) {
        u2[d] = X[1][d] - X[0][d];
        u1[d] = X[2][d] - X[1][d];
        u0[d] = X[3][d] - X[2][d];
      }
      norml3(u2, 1e-8f); norml3(u1, 1e-8f); norml3(u0, 1e-8f);
      cross3(u2, u1, n2); norml3(n2, 1e-8f);
      cross3(u1, u0, n1); norml3(n1, 1e-8f);
      float cd = dot3(n2, n1);
      cd = fminf(fmaxf(cd, -1.f + 1e-7f), 1.f - 1e-7f);
      const float sd  = dot3(u2, n1);
      const float sgn = (sd > 0.f) ? 1.f : ((sd < 0.f) ? -1.f : 0.f);
      D = sgn * acosf(cd);
    }
    df[t]     = cosf(D);
    df[t + 3] = sinf(D);
  }

  if (t == 3) {                      // rigid frame from N, CA, C
    const float* xp = xpos + ri * 12;
    float natm[3], ca[3], cc[3], e0[3], e1[3], e2[3];
    for (int d = 0; d < 3; ++d) { natm[d] = xp[d]; ca[d] = xp[3 + d]; cc[d] = xp[6 + d]; }
    for (int d = 0; d < 3; ++d) e0[d] = ca[d] - natm[d];
    norml3(e0, 1e-8f);
    for (int d = 0; d < 3; ++d) e1[d] = cc[d] - ca[d];
    const float dp = dot3(e0, e1);
    for (int d = 0; d < 3; ++d) e1[d] -= dp * e0[d];
    norml3(e1, 1e-8f);
    cross3(e0, e1, e2);
    for (int r3 = 0; r3 < 3; ++r3) {
      rot[ri * 9 + r3 * 3 + 0] = e0[r3];
      rot[ri * 9 + r3 * 3 + 1] = e1[r3];
      rot[ri * 9 + r3 * 3 + 2] = e2[r3];
    }
    for (int d = 0; d < 3; ++d) trans[ri * 3 + d] = ca[d];
  }
  __syncthreads();

  float acc = baa[t];
  for (int k = 0; k < 20; ++k) acc += aabuf[k] * Waa[k * 128 + t];
  if (aamask[ri] == 1) acc = maskemb[t];
  if (seqmask[ri] == 0) acc = pademb[t];
  float acc2 = bd[t];
  for (int k = 0; k < 6; ++k) acc2 += df[k] * Wd[k * 128 + t];
  const int  half = t >> 1;
  const float div = __expf((float)(half * 2) * (-logf(10000.f) / 128.f));
  const float pe  = (t & 1) ? cosf((float)i * div) : sinf((float)i * div);
  const float val = acc + acc2 + pe;
  s[ri * 128 + t]  = val;
  sb[ri * 128 + t] = (__bf16)val;
}

// Fused RBF featurization + z-projection WMMA + relpos epilogue.
// Block: (jtile, i, b), 128 threads = 4 waves; M tile = 16 j's, N = 128, K = 384.
__global__ __launch_bounds__(128) void k_rbf_z(
    const float* __restrict__ xpos,     // (B,L,4,3)
    const __bf16* __restrict__ WrbfT,   // (128,384) pre-transposed
    const float* __restrict__ rbf_b,    // 128
    const float* __restrict__ relw,     // (65,128)
    const float* __restrict__ relb,     // 128
    float* __restrict__ z, __bf16* __restrict__ zb, int n)
{
  __shared__ float  xi[12];
  __shared__ float  xj[16][12];
  __shared__ float  dst[16][16];        // [j'][ai*4+aj]
  __shared__ __bf16 feat[16][392];      // padded row stride

  const int jt = blockIdx.x, ii = blockIdx.y, b = blockIdx.z;
  const int t  = threadIdx.x;
  const float* xb = xpos + (long)b * n * 12;

  if (t < 12) xi[t] = xb[(long)ii * 12 + t];
  for (int idx = t; idx < 192; idx += 128)
    xj[idx / 12][idx % 12] = xb[(long)(jt * 16 + idx / 12) * 12 + (idx % 12)];
  __syncthreads();

  for (int idx = t; idx < 256; idx += 128) {
    const int jj = idx >> 4, p = idx & 15, ai = p >> 2, aj = p & 3;
    const float dx = xi[ai * 3 + 0] - xj[jj][aj * 3 + 0];
    const float dy = xi[ai * 3 + 1] - xj[jj][aj * 3 + 1];
    const float dz = xi[ai * 3 + 2] - xj[jj][aj * 3 + 2];
    dst[jj][p] = sqrtf(dx * dx + dy * dy + dz * dz + 1e-10f);
  }
  __syncthreads();

  for (int idx = t; idx < 16 * 384; idx += 128) {
    const int jj = idx / 384, f = idx % 384, p = f / 24, bin = f % 24;
    const float d = dst[jj][p] - 0.5f * (float)bin;
    feat[jj][f] = (__bf16)__expf(-4.f * d * d);    // ((d)/0.5)^2
  }
  __syncthreads();

  const int wid = t >> 5, lane = t & 31, l16 = lane & 15, khalf = lane >> 4;
  const int nb  = wid * 32;
  v8f acc[2];
  acc[0] = (v8f){0.f,0.f,0.f,0.f,0.f,0.f,0.f,0.f};
  acc[1] = acc[0];

  for (int kk = 0; kk < 12; ++kk) {
    const int k0 = kk * 32;
    ABfrag aU, bU;
    const __bf16* ar = &feat[l16][k0 + khalf * 8];
    aU.q[0] = *(const uint4*)(ar);
    aU.q[1] = *(const uint4*)(ar + 16);
#pragma unroll
    for (int sub = 0; sub < 2; ++sub) {
      const __bf16* br = WrbfT + (long)(nb + sub * 16 + l16) * 384 + k0 + khalf * 8;
      bU.q[0] = *(const uint4*)(br);
      bU.q[1] = *(const uint4*)(br + 16);
      acc[sub] = wmma_bf16(aU.v, bU.v, acc[sub]);
    }
  }

#pragma unroll
  for (int sub = 0; sub < 2; ++sub) {
    const int cz = nb + sub * 16 + l16;
#pragma unroll
    for (int v = 0; v < 8; ++v) {
      const int jj = khalf * 8 + v;
      const int j  = jt * 16 + jj;
      int rel = j - ii;
      rel = min(max(rel, -32), 32) + 32;
      const float val = acc[sub][v] + rbf_b[cz] + relw[rel * 128 + cz] + relb[cz];
      const long  o   = (((long)b * n + ii) * n + j) * 128 + cz;
      z[o]  = val;
      zb[o] = (__bf16)val;
    }
  }
}

// rotate local points into global frame: out = R p + t
__global__ __launch_bounds__(64) void k_frames_apply(
    const float* __restrict__ qp, const float* __restrict__ kvp,
    const float* __restrict__ rot, const float* __restrict__ trans,
    float* __restrict__ qpts, float* __restrict__ kvpts)
{
  const long ri = blockIdx.x;
  const int t = threadIdx.x;
  __shared__ float R[9], T[3];
  if (t < 9) R[t] = rot[ri * 9 + t];
  if (t < 3) T[t] = trans[ri * 3 + t];
  __syncthreads();
  const float* src;
  float* dstp;
  if (t < 16) { src = qp  + ri * 48  + t * 3;        dstp = qpts  + ri * 48  + t * 3; }
  else        { src = kvp + ri * 144 + (t - 16) * 3; dstp = kvpts + ri * 144 + (t - 16) * 3; }
  const float x = src[0], y = src[1], zc = src[2];
#pragma unroll
  for (int d = 0; d < 3; ++d)
    dstp[d] = R[d * 3 + 0] * x + R[d * 3 + 1] * y + R[d * 3 + 2] * zc + T[d];
}

// attention logits (+pair bias, +point-distance term, +mask) and row softmax.
__global__ __launch_bounds__(256) void k_attn(
    const float* __restrict__ qk,     // (B,H,n,n)  already scaled by 1/sqrt(96)
    const float* __restrict__ bproj,  // (B,n,n,4)
    const float* __restrict__ qpts,   // (B,n,48)
    const float* __restrict__ kvpts,  // (B,n,144)
    const float* __restrict__ hw,     // (4)
    const int* __restrict__ seqmask,  // (B,n)
    __bf16* __restrict__ a, int n)
{
  const int i = blockIdx.x % n;
  const int h = (blockIdx.x / n) % 4;
  const int b = blockIdx.x / (n * 4);
  const int j = threadIdx.x;
  __shared__ float qp[12];
  __shared__ float red[8];
  if (j < 12) qp[j] = qpts[((long)b * n + i) * 48 + h * 12 + j];
  __syncthreads();

  const float* kp = kvpts + ((long)b * n + j) * 144 + h * 36;
  float d2 = 0.f;
#pragma unroll
  for (int p = 0; p < 4; ++p)
#pragma unroll
    for (int d = 0; d < 3; ++d) {
      const float dd = qp[p * 3 + d] - kp[p * 3 + d];
      d2 += dd * dd;
    }
  const float hv  = hw[h];
  const float sp  = (hv > 20.f) ? hv : log1pf(__expf(hv));
  const float hws = sp * sqrtf(1.f / 54.f);         // sqrt(1/(3*PQK*9/2))
  const long  off = (((long)b * 4 + h) * n + i) * n + j;
  float l = qk[off]
          + 0.57735026919f * bproj[(((long)b * n + i) * n + j) * 4 + h]
          - 0.5f * d2 * hws
          + 100000.f * ((float)(seqmask[b * n + i] * seqmask[b * n + j]) - 1.f);

  const int lane = j & 31, wv = j >> 5;
  float m = l;
#pragma unroll
  for (int o = 16; o > 0; o >>= 1) m = fmaxf(m, __shfl_xor(m, o, 32));
  if (lane == 0) red[wv] = m;
  __syncthreads();
  float gm = red[0];
#pragma unroll
  for (int w = 1; w < 8; ++w) gm = fmaxf(gm, red[w]);
  const float e = __expf(l - gm);
  float ssum = e;
#pragma unroll
  for (int o = 16; o > 0; o >>= 1) ssum += __shfl_xor(ssum, o, 32);
  __syncthreads();
  if (lane == 0) red[wv] = ssum;
  __syncthreads();
  float gs = 0.f;
#pragma unroll
  for (int w = 0; w < 8; ++w) gs += red[w];
  a[off] = (__bf16)(e / gs);
}

// rotate attended points back to local frame, write bf16 coords + norms.
__global__ __launch_bounds__(32) void k_opt_finish(
    const float* __restrict__ opt,    // (B*n, H*PV*3=96)
    const float* __restrict__ rot, const float* __restrict__ trans,
    __bf16* __restrict__ concat)      // rows of 768
{
  const long ri = blockIdx.x;
  const int t = threadIdx.x;          // 32 = H*PV
  __shared__ float R[9], T[3];
  if (t < 9) R[t] = rot[ri * 9 + t];
  if (t < 3) T[t] = trans[ri * 3 + t];
  __syncthreads();
  const float* o = opt + ri * 96 + t * 3;
  const float x = o[0] - T[0], y = o[1] - T[1], zc = o[2] - T[2];
  const float v0 = R[0] * x + R[3] * y + R[6] * zc;   // R^T v
  const float v1 = R[1] * x + R[4] * y + R[7] * zc;
  const float v2 = R[2] * x + R[5] * y + R[8] * zc;
  __bf16* c = concat + ri * 768;
  c[128 + t * 3 + 0] = (__bf16)v0;
  c[128 + t * 3 + 1] = (__bf16)v1;
  c[128 + t * 3 + 2] = (__bf16)v2;
  c[224 + t] = (__bf16)sqrtf(v0 * v0 + v1 * v1 + v2 * v2 + 1e-8f);
}

// y = LayerNorm(x (+r)) * g + b ; writes f32 and optional bf16 mirror.
__global__ void k_add_ln(
    const float* __restrict__ x, const float* __restrict__ r,
    const float* __restrict__ g, const float* __restrict__ be,
    float* __restrict__ y, __bf16* __restrict__ yb, int D)
{
  __shared__ float red[16];
  const long row = blockIdx.x;
  const int  t = threadIdx.x;
  const int  lane = t & 31, wv = t >> 5, nw = blockDim.x >> 5;
  float v = x[row * D + t];
  if (r) v += r[row * D + t];
  float sum = v;
#pragma unroll
  for (int o = 16; o > 0; o >>= 1) sum += __shfl_xor(sum, o, 32);
  if (lane == 0) red[wv] = sum;
  __syncthreads();
  float tot = 0.f;
  for (int w = 0; w < nw; ++w) tot += red[w];
  const float mean = tot / (float)D;
  const float d = v - mean;
  float s2 = d * d;
#pragma unroll
  for (int o = 16; o > 0; o >>= 1) s2 += __shfl_xor(s2, o, 32);
  __syncthreads();
  if (lane == 0) red[wv] = s2;
  __syncthreads();
  float vt = 0.f;
  for (int w = 0; w < nw; ++w) vt += red[w];
  const float out = d * rsqrtf(vt / (float)D + 1e-5f) * g[t] + be[t];
  y[row * D + t] = out;
  if (yb) yb[row * D + t] = (__bf16)out;
}

// edge-transition input: [z(128) | nb_i(64) | nb_j(64)]
__global__ __launch_bounds__(256) void k_edge_concat(
    const __bf16* __restrict__ zb, const __bf16* __restrict__ nb,
    __bf16* __restrict__ econ, int n)
{
  const long row = blockIdx.x;                 // b*n*n + i*n + j
  const int  t = threadIdx.x;
  const long nn = (long)n * n;
  const long b = row / nn;
  const long ij = row - b * nn;
  const long i = ij / n, j = ij - i * n;
  __bf16 v;
  if (t < 128)      v = zb[row * 128 + t];
  else if (t < 192) v = nb[(b * n + i) * 64 + (t - 128)];
  else              v = nb[(b * n + j) * 64 + (t - 192)];
  econ[row * 256 + t] = v;
}

__global__ void k_convert(const float* __restrict__ in, __bf16* __restrict__ out, long cnt) {
  const long idx = (long)blockIdx.x * blockDim.x + threadIdx.x;
  if (idx < cnt) out[idx] = (__bf16)in[idx];
}

// fp32 (K x N) -> bf16 transposed (N x K)
__global__ void k_convT(const float* __restrict__ in, __bf16* __restrict__ out, int K, int N) {
  const int idx = blockIdx.x * blockDim.x + threadIdx.x;
  if (idx < K * N) {
    const int k = idx / N, c = idx % N;
    out[(long)c * K + k] = (__bf16)in[idx];
  }
}

// ---------------------------------------------------------------------------
// Host side
// ---------------------------------------------------------------------------
template<bool BKxN, typename OutT, int ACT>
static void launch_gemm(hipStream_t st, int M, int N, int K,
                        const __bf16* A, int lda, long sA,
                        const __bf16* B, int ldb, long sB,
                        OutT* C, int ldc, long sC,
                        const float* bias, __bf16* C2, float alpha, int batch)
{
  dim3 g((unsigned)((N + 63) / 64), (unsigned)((M + 63) / 64), (unsigned)batch);
  k_gemm<BKxN, OutT, ACT><<<g, 256, 0, st>>>(M, N, K, A, lda, sA, B, ldb, sB,
                                             C, ldc, sC, bias, C2, alpha);
}

extern "C" void kernel_launch(void* const* d_in, const int* in_sizes, int n_in,
                              void* d_out, int out_size, void* d_ws, size_t ws_size,
                              hipStream_t stream)
{
  (void)in_sizes; (void)n_in; (void)out_size; (void)ws_size;
  const int n = 256, Bb = 2, NR = Bb * 256;          // 512 residues total
  const long NN = (long)Bb * n * n;                  // 131072 pairs

  // ---- inputs (jax pytree order: dict keys sorted alphabetically) ----
  const float* x_aa  = (const float*)d_in[0];
  const float* x_pos = (const float*)d_in[1];
  const int*   aam   = (const int*)d_in[2];
  const int*   sqm   = (const int*)d_in[3];
  int cur = 4;
  auto nf  = [&]() { return (const float*)d_in[cur++]; };
  struct Lin { const float* b; const float* w; };
  auto getL = [&]() { Lin l; l.b = nf(); l.w = nf(); return l; };

  Lin aaL = getL();
  Lin dL  = getL();
  struct LayerP {
    Lin bl; bool edge; Lin e_final, e_init; const float* e_ln_b; const float* e_ln_g;
    Lin tr0, tr1; const float* hw; Lin kv, kvp; const float* ln_b; const float* ln_g;
    Lin out, post, pre, q, qp, t1, t2, t3; const float* tln_b; const float* tln_g;
  } L[6];
  for (int i = 0; i < 6; ++i) {
    L[i].bl = getL();                       // 'b'
    L[i].edge = (i != 5);                   // 'edge' (None on last layer)
    if (L[i].edge) {
      L[i].e_final = getL();                //   'final'
      L[i].e_init  = getL();                //   'init'
      L[i].e_ln_b  = nf();                  //   'ln_b'
      L[i].e_ln_g  = nf();                  //   'ln_g'
      L[i].tr0 = getL();                    //   'trunk'[0]
      L[i].tr1 = getL();                    //   'trunk'[1]
    }
    L[i].hw  = nf();                        // 'hw'
    L[i].kv  = getL(); L[i].kvp = getL();   // 'kv','kvp'
    L[i].ln_b = nf(); L[i].ln_g = nf();     // 'ln_b','ln_g'
    L[i].out = getL(); L[i].post = getL(); L[i].pre = getL();
    L[i].q = getL(); L[i].qp = getL();      // 'q','qp'
    L[i].t1 = getL(); L[i].t2 = getL(); L[i].t3 = getL();
    L[i].tln_b = nf(); L[i].tln_g = nf();
  }
  const float* mask_emb = nf();
  const float* pad_emb  = nf();
  Lin predL = getL();
  Lin rbfL  = getL();
  Lin relL  = getL();

  // ---- workspace (deterministic bump allocation) ----
  char* wsb = (char*)d_ws; size_t off = 0;
  auto alloc = [&](size_t bytes) {
    void* p = wsb + off;
    off = (off + bytes + 255) & ~(size_t)255;
    return p;
  };
  float*  s_f     = (float*)alloc((size_t)NR * 128 * 4);
  __bf16* s_b     = (__bf16*)alloc((size_t)NR * 128 * 2);
  float*  rotb    = (float*)alloc((size_t)NR * 9 * 4);
  float*  trb     = (float*)alloc((size_t)NR * 3 * 4);
  __bf16* WrbfT   = (__bf16*)alloc((size_t)384 * 128 * 2);
  float*  z_f     = (float*)alloc((size_t)NN * 128 * 4);
  __bf16* z_b     = (__bf16*)alloc((size_t)NN * 128 * 2);
  __bf16* wbuf    = (__bf16*)alloc((size_t)512 * 512 * 2);
  __bf16* qb      = (__bf16*)alloc((size_t)NR * 128 * 2);
  __bf16* kvb     = (__bf16*)alloc((size_t)NR * 256 * 2);
  float*  qpf     = (float*)alloc((size_t)NR * 48 * 4);
  float*  kvpf    = (float*)alloc((size_t)NR * 144 * 4);
  float*  qpts    = (float*)alloc((size_t)NR * 48 * 4);
  float*  kvpts   = (float*)alloc((size_t)NR * 144 * 4);
  __bf16* kvpts_b = (__bf16*)alloc((size_t)NR * 144 * 2);
  float*  bproj   = (float*)alloc((size_t)NN * 4 * 4);
  float*  qk      = (float*)alloc((size_t)Bb * 4 * n * n * 4);
  __bf16* a_b     = (__bf16*)alloc((size_t)Bb * 4 * n * n * 2);
  __bf16* concat  = (__bf16*)alloc((size_t)NR * 768 * 2);
  float*  optf    = (float*)alloc((size_t)NR * 96 * 4);
  float*  ipaout  = (float*)alloc((size_t)NR * 128 * 4);
  float*  h_f     = (float*)alloc((size_t)NR * 512 * 4);
  __bf16* h_b     = (__bf16*)alloc((size_t)NR * 512 * 2);
  __bf16* tb1     = (__bf16*)alloc((size_t)NR * 512 * 2);
  __bf16* tb2     = (__bf16*)alloc((size_t)NR * 512 * 2);
  float*  t3f     = (float*)alloc((size_t)NR * 512 * 4);
  float*  hl_f    = (float*)alloc((size_t)NR * 512 * 4);
  __bf16* hl_b    = (__bf16*)alloc((size_t)NR * 512 * 2);
  __bf16* nb_b    = (__bf16*)alloc((size_t)NR * 64 * 2);
  __bf16* econ    = (__bf16*)alloc((size_t)NN * 256 * 2);
  __bf16* e1      = (__bf16*)alloc((size_t)NN * 256 * 2);
  float*  ef      = (float*)alloc((size_t)NN * 128 * 4);

  // transpose-convert a weight into wbuf as bf16 (N x K)
  auto convT = [&](const float* w, int K, int N) {
    k_convT<<<(unsigned)(((long)K * N + 255) / 256), 256, 0, stream>>>(w, wbuf, K, N);
  };

  // ---- preprocessing ----
  k_prep<<<NR, 128, 0, stream>>>(x_aa, x_pos, aam, sqm, aaL.w, aaL.b, dL.w, dL.b,
                                 mask_emb, pad_emb, s_f, s_b, rotb, trb, n);
  k_convT<<<(384 * 128 + 255) / 256, 256, 0, stream>>>(rbfL.w, WrbfT, 384, 128);
  {
    dim3 gz(16, (unsigned)n, (unsigned)Bb);
    k_rbf_z<<<gz, 128, 0, stream>>>(x_pos, WrbfT, rbfL.b, relL.w, relL.b, z_f, z_b, n);
  }

  const float qkscale = 1.0f / sqrtf(96.0f);
  const long nn = (long)n * n;

  for (int li = 0; li < 6; ++li) {
    const LayerP& Ly = L[li];

    // ---- IPA projections (weights pre-transposed -> async-capable path) ----
    convT(Ly.q.w, 128, 128);
    launch_gemm<false, __bf16, 0>(stream, NR, 128, 128, s_b, 128, 0, wbuf, 128, 0,
                                  qb, 128, 0, Ly.q.b, nullptr, 1.f, 1);
    convT(Ly.kv.w, 128, 256);
    launch_gemm<false, __bf16, 0>(stream, NR, 256, 128, s_b, 128, 0, wbuf, 128, 0,
                                  kvb, 256, 0, Ly.kv.b, nullptr, 1.f, 1);
    convT(Ly.qp.w, 128, 48);
    launch_gemm<false, float, 0>(stream, NR, 48, 128, s_b, 128, 0, wbuf, 128, 0,
                                 qpf, 48, 0, Ly.qp.b, nullptr, 1.f, 1);
    convT(Ly.kvp.w, 128, 144);
    launch_gemm<false, float, 0>(stream, NR, 144, 128, s_b, 128, 0, wbuf, 128, 0,
                                 kvpf, 144, 0, Ly.kvp.b, nullptr, 1.f, 1);
    k_frames_apply<<<NR, 64, 0, stream>>>(qpf, kvpf, rotb, trb, qpts, kvpts);
    k_convert<<<(NR * 144 + 255) / 256, 256, 0, stream>>>(kvpts, kvpts_b, (long)NR * 144);

    // pair bias projection over z
    convT(Ly.bl.w, 128, 4);
    launch_gemm<false, float, 0>(stream, (int)NN, 4, 128, z_b, 128, 0, wbuf, 128, 0,
                                 bproj, 4, 0, Ly.bl.b, nullptr, 1.f, 1);

    // q @ k^T per (b,h), alpha = 1/sqrt(3C)
    for (int b2 = 0; b2 < Bb; ++b2)
      launch_gemm<false, float, 0>(stream, n, n, 32,
          qb  + (long)b2 * n * 128, 128, 32,
          kvb + (long)b2 * n * 256, 256, 64,
          qk  + (long)b2 * 4 * nn,  n,  nn,
          nullptr, nullptr, qkscale, 4);

    k_attn<<<Bb * 4 * n, 256, 0, stream>>>(qk, bproj, qpts, kvpts, Ly.hw, sqm, a_b, n);

    // o = a @ v  -> concat[:,0:128]
    for (int b2 = 0; b2 < Bb; ++b2)
      launch_gemm<true, __bf16, 0>(stream, n, 32, n,
          a_b + (long)b2 * 4 * nn, n, nn,
          kvb + (long)b2 * n * 256 + 32, 256, 64,
          concat + (long)b2 * n * 768, 768, 32,
          nullptr, nullptr, 1.f, 4);

    // o_pt = a @ v_pts (N=24) -> optf
    for (int b2 = 0; b2 < Bb; ++b2)
      launch_gemm<true, float, 0>(stream, n, 24, n,
          a_b + (long)b2 * 4 * nn, n, nn,
          kvpts_b + (long)b2 * n * 144 + 12, 144, 36,
          optf + (long)b2 * n * 96, 96, 24,
          nullptr, nullptr, 1.f, 4);
    k_opt_finish<<<NR, 32, 0, stream>>>(optf, rotb, trb, concat);

    // o_pair = a @ z (per (b,i): 4x256x128) -> concat[:,256:768]
    for (int b2 = 0; b2 < Bb; ++b2)
      launch_gemm<true, __bf16, 0>(stream, 4, 128, n,
          a_b + (long)b2 * 4 * nn, (int)nn, n,
          z_b + (long)b2 * nn * 128, 128, (long)n * 128,
          concat + (long)b2 * n * 768 + 256, 128, 768,
          nullptr, nullptr, 1.f, n);

    // IPA output linear + residual layernorm
    convT(Ly.out.w, 768, 128);
    launch_gemm<false, float, 0>(stream, NR, 128, 768, concat, 768, 0, wbuf, 768, 0,
                                 ipaout, 128, 0, Ly.out.b, nullptr, 1.f, 1);
    k_add_ln<<<NR, 128, 0, stream>>>(ipaout, s_f, Ly.ln_g, Ly.ln_b, s_f, s_b, 128);

    // ---- transition MLP ----
    convT(Ly.pre.w, 128, 512);
    launch_gemm<false, float, 0>(stream, NR, 512, 128, s_b, 128, 0, wbuf, 128, 0,
                                 h_f, 512, 0, Ly.pre.b, h_b, 1.f, 1);
    convT(Ly.t1.w, 512, 512);
    launch_gemm<false, __bf16, 1>(stream, NR, 512, 512, h_b, 512, 0, wbuf, 512, 0,
                                  tb1, 512, 0, Ly.t1.b, nullptr, 1.f, 1);
    convT(Ly.t2.w, 512, 512);
    launch_gemm<false, __bf16, 1>(stream, NR, 512, 512, tb1, 512, 0, wbuf, 512, 0,
                                  tb2, 512, 0, Ly.t2.b, nullptr, 1.f, 1);
    convT(Ly.t3.w, 512, 512);
    launch_gemm<false, float, 0>(stream, NR, 512, 512, tb2, 512, 0, wbuf, 512, 0,
                                 t3f, 512, 0, Ly.t3.b, nullptr, 1.f, 1);
    k_add_ln<<<NR, 512, 0, stream>>>(t3f, h_f, Ly.tln_g, Ly.tln_b, hl_f, hl_b, 512);
    convT(Ly.post.w, 512, 128);
    launch_gemm<false, float, 0>(stream, NR, 128, 512, hl_b, 512, 0, wbuf, 512, 0,
                                 s_f, 128, 0, Ly.post.b, s_b, 1.f, 1);

    // ---- edge transition ----
    if (Ly.edge) {
      convT(Ly.e_init.w, 128, 64);
      launch_gemm<false, __bf16, 0>(stream, NR, 64, 128, s_b, 128, 0, wbuf, 128, 0,
                                    nb_b, 64, 0, Ly.e_init.b, nullptr, 1.f, 1);
      k_edge_concat<<<(unsigned)NN, 256, 0, stream>>>(z_b, nb_b, econ, n);
      convT(Ly.tr0.w, 256, 256);
      launch_gemm<false, __bf16, 1>(stream, (int)NN, 256, 256, econ, 256, 0, wbuf, 256, 0,
                                    e1, 256, 0, Ly.tr0.b, nullptr, 1.f, 1);
      convT(Ly.tr1.w, 256, 256);
      launch_gemm<false, __bf16, 1>(stream, (int)NN, 256, 256, e1, 256, 0, wbuf, 256, 0,
                                    econ, 256, 0, Ly.tr1.b, nullptr, 1.f, 1);
      convT(Ly.e_final.w, 256, 128);
      launch_gemm<false, float, 0>(stream, (int)NN, 128, 256, econ, 256, 0, wbuf, 256, 0,
                                   ef, 128, 0, Ly.e_final.b, nullptr, 1.f, 1);
      k_add_ln<<<(unsigned)NN, 128, 0, stream>>>(ef, nullptr, Ly.e_ln_g, Ly.e_ln_b,
                                                 z_f, z_b, 128);
    }
  }

  // ---- final prediction head ----
  convT(predL.w, 128, 20);
  launch_gemm<false, float, 0>(stream, NR, 20, 128, s_b, 128, 0, wbuf, 128, 0,
                               (float*)d_out, 20, 0, predL.b, nullptr, 1.f, 1);
}
